// DecentralPlannerGATNet_2199023256009
// MI455X (gfx1250) — matrix operand order repaired and checked by reading, not compiled, not run on hardware
//
#include <hip/hip_runtime.h>
#include <hip/hip_bf16.h>
#include <math.h>

typedef __attribute__((ext_vector_type(16))) _Float16 v16h;
typedef __attribute__((ext_vector_type(8)))  float    v8f;

// ---------------------------------------------------------------------------
// Direct 3x3 same-padding conv at one output position (NCHW, OIHW weights)
// ---------------------------------------------------------------------------
__device__ __forceinline__ float conv_at(const float* __restrict__ X,
                                         const float* __restrict__ Wt,
                                         int n, int co, int h, int w,
                                         int Cin, int H, int W)
{
  float acc = 0.0f;
  const float* wbase = Wt + (size_t)co * Cin * 9;
  const float* xbase = X + (size_t)n * Cin * H * W;
  for (int ci = 0; ci < Cin; ++ci) {
    const float* xc = xbase + (size_t)ci * H * W;
    const float* wc = wbase + ci * 9;
#pragma unroll
    for (int kh = 0; kh < 3; ++kh) {
      int hh = h + kh - 1;
      if (hh < 0 || hh >= H) continue;
#pragma unroll
      for (int kw = 0; kw < 3; ++kw) {
        int ww = w + kw - 1;
        if (ww < 0 || ww >= W) continue;
        acc = fmaf(xc[hh * W + ww], wc[kh * 3 + kw], acc);
      }
    }
  }
  return acc;
}

// ---------------------------------------------------------------------------
// Pass 1: per-(channel, node-chunk) partial sum/sumsq of conv output (pre-pool)
// Deterministic: fixed thread->element assignment + LDS tree reduce.
// grid = Cout * SPLIT, block = 256
// ---------------------------------------------------------------------------
__global__ __launch_bounds__(256) void conv_stats_partial(
    const float* __restrict__ X, const float* __restrict__ Wt,
    const float* __restrict__ bconv, float* __restrict__ part,
    int N, int Cin, int H, int W, int SPLIT)
{
  int c = blockIdx.x / SPLIT;
  int s = blockIdx.x % SPLIT;
  int per = (N + SPLIT - 1) / SPLIT;
  long n0 = (long)s * per;
  long n1 = n0 + per; if (n1 > N) n1 = N;
  int HW = H * W;
  float bc = bconv[c];
  float sum = 0.0f, ssum = 0.0f;
  for (long idx = n0 * HW + threadIdx.x; idx < n1 * HW; idx += blockDim.x) {
    int n = (int)(idx / HW);
    int p = (int)(idx % HW);
    float y = conv_at(X, Wt, n, c, p / W, p % W, Cin, H, W) + bc;
    sum += y;
    ssum = fmaf(y, y, ssum);
  }
  __shared__ float s1[256], s2[256];
  s1[threadIdx.x] = sum; s2[threadIdx.x] = ssum;
  __syncthreads();
  for (int st = 128; st > 0; st >>= 1) {
    if ((int)threadIdx.x < st) {
      s1[threadIdx.x] += s1[threadIdx.x + st];
      s2[threadIdx.x] += s2[threadIdx.x + st];
    }
    __syncthreads();
  }
  if (threadIdx.x == 0) {
    part[(size_t)blockIdx.x * 2]     = s1[0];
    part[(size_t)blockIdx.x * 2 + 1] = s2[0];
  }
}

// Reduce SPLIT partials per channel -> affine BN coefficients (a, b)
__global__ void stats_reduce(const float* __restrict__ part, float* __restrict__ stats,
                             int Cout, int SPLIT, float inv_count,
                             const float* __restrict__ gamma, const float* __restrict__ beta)
{
  int c = blockIdx.x * blockDim.x + threadIdx.x;
  if (c >= Cout) return;
  float s = 0.0f, ss = 0.0f;
  for (int i = 0; i < SPLIT; ++i) {
    s  += part[(size_t)(c * SPLIT + i) * 2];
    ss += part[(size_t)(c * SPLIT + i) * 2 + 1];
  }
  float mean = s * inv_count;
  float var  = ss * inv_count - mean * mean;
  float inv  = rsqrtf(var + 1e-5f);
  float a = gamma[c] * inv;
  stats[c * 2]     = a;
  stats[c * 2 + 1] = beta[c] - mean * a;
}

// Pass 2: fused conv + BN + ReLU + optional 2x2 maxpool (relu(max)=max(relu))
__global__ __launch_bounds__(256) void conv_apply(
    const float* __restrict__ X, const float* __restrict__ Wt,
    const float* __restrict__ bconv, const float* __restrict__ stats,
    float* __restrict__ Y, int N, int Cin, int Cout, int H, int W, int pool)
{
  int Ho = pool ? H / 2 : H;
  int Wo = pool ? W / 2 : W;
  long total = (long)N * Cout * Ho * Wo;
  long idx = (long)blockIdx.x * blockDim.x + threadIdx.x;
  if (idx >= total) return;
  int wo = (int)(idx % Wo); long t = idx / Wo;
  int ho = (int)(t % Ho);   t /= Ho;
  int c  = (int)(t % Cout);
  int n  = (int)(t / Cout);
  float a  = stats[c * 2];
  float b2 = stats[c * 2 + 1];
  float bc = bconv[c];
  float best;
  if (pool) {
    best = 0.0f; // relu outputs are >= 0
#pragma unroll
    for (int dh = 0; dh < 2; ++dh)
#pragma unroll
      for (int dw = 0; dw < 2; ++dw) {
        float y = conv_at(X, Wt, n, c, ho * 2 + dh, wo * 2 + dw, Cin, H, W) + bc;
        y = fmaxf(fmaf(a, y, b2), 0.0f);
        best = fmaxf(best, y);
      }
  } else {
    float y = conv_at(X, Wt, n, c, ho, wo, Cin, H, W) + bc;
    best = fmaxf(fmaf(a, y, b2), 0.0f);
  }
  Y[((size_t)n * Cout + c) * (size_t)(Ho * Wo) + ho * Wo + wo] = best;
}

// ---------------------------------------------------------------------------
// WMMA fragment packing (f32 memory -> f16 fragments, b128 loads)
// ---------------------------------------------------------------------------
__device__ __forceinline__ v16h pack_a(const float* __restrict__ arow, int k0, int kh)
{
  // 16-bit A 16x32 layout: halves 0..7 -> K = k0+8*kh+0..7 ; halves 8..15 -> +16
  const float4 a0 = *(const float4*)(arow + k0 + 8 * kh);
  const float4 a1 = *(const float4*)(arow + k0 + 8 * kh + 4);
  const float4 a2 = *(const float4*)(arow + k0 + 16 + 8 * kh);
  const float4 a3 = *(const float4*)(arow + k0 + 16 + 8 * kh + 4);
  v16h a;
  a[0]=(_Float16)a0.x;  a[1]=(_Float16)a0.y;  a[2]=(_Float16)a0.z;  a[3]=(_Float16)a0.w;
  a[4]=(_Float16)a1.x;  a[5]=(_Float16)a1.y;  a[6]=(_Float16)a1.z;  a[7]=(_Float16)a1.w;
  a[8]=(_Float16)a2.x;  a[9]=(_Float16)a2.y;  a[10]=(_Float16)a2.z; a[11]=(_Float16)a2.w;
  a[12]=(_Float16)a3.x; a[13]=(_Float16)a3.y; a[14]=(_Float16)a3.z; a[15]=(_Float16)a3.w;
  return a;
}

__device__ __forceinline__ v16h pack_b16(const float* __restrict__ p, int Nn, int kh)
{
  // B layout: lane -> n = lane&15; halves 0..7 -> K = 8*kh+0..7 ; halves 8..15 -> +16 (stride Nn)
  v16h b;
#pragma unroll
  for (int s = 0; s < 8; ++s) {
    b[s]     = (_Float16)p[(size_t)(8 * kh + s) * Nn];
    b[s + 8] = (_Float16)p[(size_t)(16 + 8 * kh + s) * Nn];
  }
  return b;
}

// ---------------------------------------------------------------------------
// WMMA GEMM: C[M,Nn] = A[M,K] * B[K,Nn] (+bias) (optional relu). f32 memory,
// f16 operands, f32 accumulate. One wave computes a 16x32 C tile (two WMMA
// accumulators share one A fragment). K templated (128/512) => full unroll.
// Requires M%16==0, Nn%32==0.
// ---------------------------------------------------------------------------
template<int K>
__global__ __launch_bounds__(256) void gemm_wmma(
    const float* __restrict__ A, const float* __restrict__ B,
    const float* __restrict__ bias, float* __restrict__ C,
    int M, int Nn, int doRelu, int nTiles)
{
  int lane = threadIdx.x & 31;
  int wave = threadIdx.x >> 5;
  int tile = blockIdx.x * (blockDim.x >> 5) + wave;
  if (tile >= nTiles) return;          // whole-wave exit: EXEC stays all-ones
  int tilesN = Nn >> 5;                // 32-wide N tiles
  int tm = tile / tilesN;
  int tn = tile - tm * tilesN;

  int m  = tm * 16 + (lane & 15);      // A row for this lane
  int kh = lane >> 4;                  // K-half select (ISA 16-bit A 16x32 layout)
  const float* arow  = A + (size_t)m * K;
  const float* bbase = B + tn * 32 + (lane & 15);

  v8f acc0 = {};
  v8f acc1 = {};
#pragma unroll
  for (int k0 = 0; k0 < K; k0 += 32) {
    v16h a  = pack_a(arow, k0, kh);
    const float* bp = bbase + (size_t)k0 * Nn;   // row k0, column n
    v16h b0 = pack_b16(bp, Nn, kh);
    v16h b1 = pack_b16(bp + 16, Nn, kh);
    acc0 = __builtin_amdgcn_wmma_f32_16x16x32_f16(
        false, a, false, b0, (short)0, acc0, false, false);
    acc1 = __builtin_amdgcn_wmma_f32_16x16x32_f16(
        false, a, false, b1, (short)0, acc1, false, false);
  }
  // C/D layout: lane -> n = lane&15; VGPR r -> m = r + 8*(lane>>4)
  int cn = tn * 32 + (lane & 15);
  float bv0 = bias ? bias[cn]      : 0.0f;
  float bv1 = bias ? bias[cn + 16] : 0.0f;
#pragma unroll
  for (int r = 0; r < 8; ++r) {
    int cm = tm * 16 + r + 8 * kh;
    float v0 = acc0[r] + bv0;
    float v1 = acc1[r] + bv1;
    if (doRelu) { v0 = fmaxf(v0, 0.0f); v1 = fmaxf(v1, 0.0f); }
    C[(size_t)cm * Nn + cn]      = v0;
    C[(size_t)cm * Nn + cn + 16] = v1;
  }
}

// ---------------------------------------------------------------------------
// GAT helpers
// ---------------------------------------------------------------------------
__device__ __forceinline__ void atomicMaxFloat(float* addr, float val) {
  int* ai = (int*)addr;
  int cur = *ai;
  while (__int_as_float(cur) < val) {
    int prev = atomicCAS(ai, cur, __float_as_int(val));
    if (prev == cur) break;
    cur = prev;
  }
}

// a_s[n,h] = <xl[n,h,:], att_src[h,:]>, same for a_d
__global__ void gat_att(const float* __restrict__ XL, const float* __restrict__ att_src,
                        const float* __restrict__ att_dst,
                        float* __restrict__ asb, float* __restrict__ adb, int N)
{
  int i = blockIdx.x * blockDim.x + threadIdx.x;
  if (i >= N * 4) return;
  int n = i >> 2, h = i & 3;
  const float* x = XL + (size_t)n * 512 + h * 128;
  const float* s = att_src + h * 128;
  const float* d = att_dst + h * 128;
  float vs = 0.0f, vd = 0.0f;
  for (int o = 0; o < 128; ++o) { float xv = x[o]; vs = fmaf(xv, s[o], vs); vd = fmaf(xv, d[o], vd); }
  asb[i] = vs; adb[i] = vd;
}

__global__ void gat_init(float* __restrict__ agg, float* __restrict__ mbuf,
                         float* __restrict__ dbuf, long naggr, int nh)
{
  long i = (long)blockIdx.x * blockDim.x + threadIdx.x;
  if (i < naggr) agg[i] = 0.0f;
  if (i < nh) { mbuf[i] = -3.0e38f; dbuf[i] = 0.0f; }
}

// edge pass 1: e = leaky_relu(a_s[src] + a_d[dst]); segment-max into mbuf[dst]
__global__ void gat_edge1(const int* __restrict__ ei, const float* __restrict__ asb,
                          const float* __restrict__ adb, float* __restrict__ ebuf,
                          float* __restrict__ mbuf, int E, int N)
{
  long i = (long)blockIdx.x * blockDim.x + threadIdx.x;
  long tot = (long)(E + N) * 4;
  if (i >= tot) return;
  int h = (int)(i & 3);
  long e = i >> 2;
  int src = (e < E) ? ei[e]     : (int)(e - E);
  int dst = (e < E) ? ei[E + e] : (int)(e - E);
  float v = asb[src * 4 + h] + adb[dst * 4 + h];
  v = (v > 0.0f) ? v : 0.2f * v;
  ebuf[i] = v;
  atomicMaxFloat(&mbuf[dst * 4 + h], v);
}

// edge pass 2: ee = exp(e - m[dst]); denom[dst] += ee
__global__ void gat_edge2(const int* __restrict__ ei, float* __restrict__ ebuf,
                          const float* __restrict__ mbuf, float* __restrict__ dbuf,
                          int E, int N)
{
  long i = (long)blockIdx.x * blockDim.x + threadIdx.x;
  long tot = (long)(E + N) * 4;
  if (i >= tot) return;
  int h = (int)(i & 3);
  long e = i >> 2;
  int dst = (e < E) ? ei[E + e] : (int)(e - E);
  float ee = expf(ebuf[i] - mbuf[dst * 4 + h]);
  ebuf[i] = ee;
  atomicAdd(&dbuf[dst * 4 + h], ee);
}

// edge pass 3: out[dst,h,:] += alpha * xl[src,h,:]; 32 threads per (edge,head)
__global__ void gat_edge3(const int* __restrict__ ei, const float* __restrict__ ebuf,
                          const float* __restrict__ dbuf, const float* __restrict__ XL,
                          float* __restrict__ agg, int E, int N)
{
  long i = (long)blockIdx.x * blockDim.x + threadIdx.x;
  long tot = (long)(E + N) * 4 * 32;
  if (i >= tot) return;
  int sub = (int)(i & 31);
  long t = i >> 5;
  int h = (int)(t & 3);
  long e = t >> 2;
  int src = (e < E) ? ei[e]     : (int)(e - E);
  int dst = (e < E) ? ei[E + e] : (int)(e - E);
  float alpha = ebuf[t] / (dbuf[dst * 4 + h] + 1e-16f);
  const float* xs = XL + (size_t)src * 512 + h * 128 + sub * 4;
  float* od = agg + (size_t)dst * 512 + h * 128 + sub * 4;
  atomicAdd(&od[0], alpha * xs[0]);
  atomicAdd(&od[1], alpha * xs[1]);
  atomicAdd(&od[2], alpha * xs[2]);
  atomicAdd(&od[3], alpha * xs[3]);
}

__global__ void bias_relu(const float* __restrict__ in, const float* __restrict__ bias,
                          float* __restrict__ out, long total, int Nfeat)
{
  long i = (long)blockIdx.x * blockDim.x + threadIdx.x;
  if (i >= total) return;
  int j = (int)(i % Nfeat);
  out[i] = fmaxf(in[i] + bias[j], 0.0f);
}

// final 128 -> 5 head (tiny)
__global__ void final_fc(const float* __restrict__ Y1, const float* __restrict__ W2,
                         const float* __restrict__ b2, float* __restrict__ out, int N)
{
  int i = blockIdx.x * blockDim.x + threadIdx.x;
  if (i >= N * 5) return;
  int n = i / 5, j = i % 5;
  const float* y = Y1 + (size_t)n * 128;
  float acc = b2[j];
  for (int k = 0; k < 128; ++k) acc = fmaf(y[k], W2[k * 5 + j], acc);
  out[i] = acc;
}

// ---------------------------------------------------------------------------
static inline unsigned cdivu(long a, long b) { return (unsigned)((a + b - 1) / b); }

extern "C" void kernel_launch(void* const* d_in, const int* in_sizes, int n_in,
                              void* d_out, int out_size, void* d_ws, size_t ws_size,
                              hipStream_t stream)
{
  (void)n_in; (void)out_size; (void)ws_size;
  const int N = in_sizes[0] / (3 * 11 * 11);
  const int E = in_sizes[1] / 2;

  const float* x_in = (const float*)d_in[0];
  const int*   ei   = (const int*)d_in[1];
  const float *cw[5], *cb[5], *cg[5], *cbe[5];
  for (int l = 0; l < 5; ++l) {
    cw[l]  = (const float*)d_in[2 + 4 * l];
    cb[l]  = (const float*)d_in[3 + 4 * l];
    cg[l]  = (const float*)d_in[4 + 4 * l];
    cbe[l] = (const float*)d_in[5 + 4 * l];
  }
  const float* mlpW = (const float*)d_in[22];
  const float* mlpB = (const float*)d_in[23];
  const float *gW[3], *gAs[3], *gAd[3], *gB[3];
  for (int g = 0; g < 3; ++g) {
    gW[g]  = (const float*)d_in[24 + 4 * g];
    gAs[g] = (const float*)d_in[25 + 4 * g];
    gAd[g] = (const float*)d_in[26 + 4 * g];
    gB[g]  = (const float*)d_in[27 + 4 * g];
  }
  const float* aW1 = (const float*)d_in[36];
  const float* aB1 = (const float*)d_in[37];
  const float* aW2 = (const float*)d_in[38];
  const float* aB2 = (const float*)d_in[39];
  float* out = (float*)d_out;

  // workspace layout
  float* ws = (float*)d_ws;
  size_t off = 0;
  float* bufA = ws + off; off += (size_t)N * 800;       // peak: post-pool conv0 (N,32,5,5)
  float* bufB = ws + off; off += (size_t)N * 800;
  float* xlb  = ws + off; off += (size_t)N * 512;       // GAT linear output
  float* ebuf = ws + off; off += (size_t)(E + N) * 4;   // per-edge attention
  float* mbuf = ws + off; off += (size_t)N * 4;
  float* dbuf = ws + off; off += (size_t)N * 4;
  float* asb  = ws + off; off += (size_t)N * 4;
  float* adb  = ws + off; off += (size_t)N * 4;
  float* part = ws + off; off += 128 * 64 * 2;
  float* stats= ws + off; off += 256;

  const int SPLIT = 64;

  auto conv_layer = [&](const float* X, float* Y, int l,
                        int Cin, int Cout, int H, int W, int pool) {
    conv_stats_partial<<<dim3(Cout * SPLIT), dim3(256), 0, stream>>>(
        X, cw[l], cb[l], part, N, Cin, H, W, SPLIT);
    float inv_count = 1.0f / ((float)N * H * W);
    stats_reduce<<<dim3(1), dim3(128), 0, stream>>>(
        part, stats, Cout, SPLIT, inv_count, cg[l], cbe[l]);
    int Ho = pool ? H / 2 : H, Wo = pool ? W / 2 : W;
    long total = (long)N * Cout * Ho * Wo;
    conv_apply<<<dim3(cdivu(total, 256)), dim3(256), 0, stream>>>(
        X, cw[l], cb[l], stats, Y, N, Cin, Cout, H, W, pool);
  };

  auto gemm = [&](const float* A, const float* B, const float* bias, float* C,
                  int M, int Nc, int K, int relu) {
    int tiles = (M / 16) * (Nc / 32);
    long threads = (long)tiles * 32;
    dim3 grid(cdivu(threads, 256)), blk(256);
    if (K == 128)
      gemm_wmma<128><<<grid, blk, 0, stream>>>(A, B, bias, C, M, Nc, relu, tiles);
    else
      gemm_wmma<512><<<grid, blk, 0, stream>>>(A, B, bias, C, M, Nc, relu, tiles);
  };

  auto gat = [&](const float* X, float* Yout, int g, int din) {
    gemm(X, gW[g], nullptr, xlb, N, 512, din, 0);
    int nh = N * 4;
    gat_att<<<dim3(cdivu(nh, 256)), dim3(256), 0, stream>>>(xlb, gAs[g], gAd[g], asb, adb, N);
    long naggr = (long)N * 512;
    gat_init<<<dim3(cdivu(naggr, 256)), dim3(256), 0, stream>>>(Yout, mbuf, dbuf, naggr, nh);
    long et4 = (long)(E + N) * 4;
    gat_edge1<<<dim3(cdivu(et4, 256)), dim3(256), 0, stream>>>(ei, asb, adb, ebuf, mbuf, E, N);
    gat_edge2<<<dim3(cdivu(et4, 256)), dim3(256), 0, stream>>>(ei, ebuf, mbuf, dbuf, E, N);
    long e32 = et4 * 32;
    gat_edge3<<<dim3(cdivu(e32, 256)), dim3(256), 0, stream>>>(ei, ebuf, dbuf, xlb, Yout, E, N);
    bias_relu<<<dim3(cdivu(naggr, 256)), dim3(256), 0, stream>>>(Yout, gB[g], Yout, naggr, 512);
  };

  // CNN encoder: (3,11,11)->pool(5,5)->(32)->(64)pool(2,2)->(64)->(128)pool(1,1)
  conv_layer(x_in, bufA, 0,  3,  32, 11, 11, 1);
  conv_layer(bufA, bufB, 1, 32,  32,  5,  5, 0);
  conv_layer(bufB, bufA, 2, 32,  64,  5,  5, 1);
  conv_layer(bufA, bufB, 3, 64,  64,  2,  2, 0);
  conv_layer(bufB, bufA, 4, 64, 128,  2,  2, 1);   // -> (N,128)

  // MLP 128 -> 128 (relu)
  gemm(bufA, mlpW, mlpB, bufB, N, 128, 128, 1);

  // 3 GAT layers (relu applied inside)
  gat(bufB, bufA, 0, 128);
  gat(bufA, bufB, 1, 512);
  gat(bufB, bufA, 2, 512);

  // action head: 512 -> 128 (relu), 128 -> 5
  gemm(bufA, aW1, aB1, bufB, N, 128, 512, 1);
  final_fc<<<dim3(cdivu((long)N * 5, 256)), dim3(256), 0, stream>>>(bufB, aW2, aB2, out, N);
}
